// Diagnet_30210799960121
// MI455X (gfx1250) — matrix operand order repaired and checked by reading, not compiled, not run on hardware
//
#include <hip/hip_runtime.h>
#include <hip/hip_bf16.h>
#include <stdint.h>

// ---------------------------------------------------------------------------
// Diagnet on MI455X (gfx1250):
//   u = X@W_ih^T (137 GFLOP, dominant)  ->  h = |u_t + hh*h| scan (serial in S)
//   ->  Y = h_final@W_ho^T + b  (0.13 GFLOP).
// Fused GEMM+scan: h stays in VGPRs for all 1024 steps, W_ih tile lives in LDS
// as bf16 (staged once, reused 1024x), X slices stream HBM->LDS via the CDNA5
// async-to-LDS engine (ASYNCcnt) double-buffered so the copy of step s+1 runs
// under the WMMAs of step s. bf16 WMMA 16x16x32 w/ fp32 accumulate = 8x the
// per-instruction FLOPs of the f32 16x16x4 path; inputs are well-conditioned
// (X~N(0,1), W ~ 1/sqrt(I)) so bf16 operands cost ~0.3% relative error.
// ---------------------------------------------------------------------------

typedef __attribute__((ext_vector_type(16))) __bf16 v16bf;
typedef __attribute__((ext_vector_type(8)))  float  v8f;

union FragBF {
  v16bf    bf;
  uint32_t u[8];
  uint4    q[2];
};

static constexpr int S_ = 1024, B_ = 64, I_ = 512, H_ = 2048, O_ = 512;
static constexpr int HT = 32;   // h-tile per workgroup (2 waves x 16)

// single-instruction RNE pack: two fp32 -> packed bf16x2
__device__ __forceinline__ uint32_t cvt_pk_bf16(float lo, float hi) {
  uint32_t r;
  asm("v_cvt_pk_bf16_f32 %0, %1, %2" : "=v"(r) : "v"(lo), "v"(hi));
  return r;
}

// CDNA5 async copy: 16 bytes/lane, global (SGPR base + 32-bit voffset) -> LDS
__device__ __forceinline__ void async_g2l_b128(uint32_t lds_byte_addr,
                                               uint32_t voff, uint64_t gbase) {
  asm volatile("global_load_async_to_lds_b128 %0, %1, %2"
               :: "v"(lds_byte_addr), "v"(voff), "s"(gbase) : "memory");
}

__device__ __forceinline__ void wait_async0() {
  asm volatile("s_wait_asynccnt 0x0" ::: "memory");
}

// ---------------------------------------------------------------------------
// Phase 1: fused  u_t = X[t] @ W_ih^T  +  h = |u_t + hh*h|  scan over S.
// Grid (H/32, B/16), 64 threads (2 waves); wave w owns N-subtile [16w,16w+16).
// LDS: 32 KB bf16 W tile + 2 x 32 KB fp32 X slice double buffer = 96 KB.
// ---------------------------------------------------------------------------
__global__ void __launch_bounds__(64)
diagnet_scan(const float* __restrict__ X, const float* __restrict__ W_ih,
             const float* __restrict__ hh, float* __restrict__ hfin) {
  __shared__ __align__(16) uint32_t s_w[HT * I_ / 2];   // bf16 W tile
  __shared__ __align__(16) float    s_x[2][16 * I_];    // fp32 X double buffer

  const int tid  = threadIdx.x;
  const int wave = tid >> 5;
  const int lane = tid & 31;
  const int ln16 = lane & 15;
  const int hf   = lane >> 4;          // lane half (ISA fragment layouts)
  const int h0   = blockIdx.x * HT;
  const int b0   = blockIdx.y * 16;

  // Stage W_ih[h0:h0+32, :] as bf16 into LDS (once, reused for 1024 steps)
  {
    const float4* wg = (const float4*)(W_ih + (size_t)h0 * I_);
    for (int j = tid; j < HT * I_ / 4; j += 64) {
      float4 f = wg[j];
      s_w[2 * j]     = cvt_pk_bf16(f.x, f.y);
      s_w[2 * j + 1] = cvt_pk_bf16(f.z, f.w);
    }
  }

  // Kick off async stage of slice s=0 into buffer 0 (32 KB, contiguous rows)
  {
    const uint64_t g  = (uint64_t)(uintptr_t)(X + (size_t)b0 * I_);
    const uint32_t l0 = (uint32_t)(uintptr_t)&s_x[0][0];
#pragma unroll 4
    for (int j = 0; j < 32; ++j) {
      const uint32_t off = (uint32_t)((j * 64 + tid) * 16);
      async_g2l_b128(l0 + off, off, g);
    }
  }

  const float hhv = hh[h0 + wave * 16 + ln16];  // recurrence diag: N-only
  const uint32_t* wrow  = s_w + (size_t)(wave * 16 + ln16) * (I_ / 2);
  const float*    xrow0 = &s_x[0][(size_t)ln16 * I_];
  const float*    xrow1 = &s_x[1][(size_t)ln16 * I_];

  v8f hstate = {};

  for (int s = 0; s < S_; ++s) {
    wait_async0();       // this wave's async writes landed in LDS
    __syncthreads();     // everyone's writes visible; prev buffer reads done

    // Issue async stage of slice s+1 into the other buffer: it streams from
    // L2/HBM concurrently with this step's WMMAs (off the critical path).
    if (s + 1 < S_) {
      const uint64_t g  = (uint64_t)(uintptr_t)(X + ((size_t)(s + 1) * B_ + b0) * I_);
      const uint32_t lb = (uint32_t)(uintptr_t)&s_x[(s + 1) & 1][0];
#pragma unroll 4
      for (int j = 0; j < 32; ++j) {
        const uint32_t off = (uint32_t)((j * 64 + tid) * 16);
        async_g2l_b128(lb + off, off, g);
      }
    }

    const float* xr = (s & 1) ? xrow1 : xrow0;
    v8f acc = {};
#pragma unroll
    for (int kb = 0; kb < I_ / 32; ++kb) {
      // B frag (32x16 bf16): lane half hf covers K = kb*32+8*hf .. +7 and kb*32+16+8*hf .. +7
      FragBF bfr;
      const uint32_t* bp = wrow + kb * 16 + hf * 4;
      bfr.q[0] = *(const uint4*)(bp);
      bfr.q[1] = *(const uint4*)(bp + 8);
      // A frag (16x32 bf16): row m=ln16; fp32 groups at K+8*hf and K+16+8*hf,
      // packed with v_cvt_pk_bf16_f32 (co-executes with the XDL pipe).
      const float* ap = xr + kb * 32 + 8 * hf;
      float4 f0 = *(const float4*)(ap);
      float4 f1 = *(const float4*)(ap + 4);
      float4 g0 = *(const float4*)(ap + 16);
      float4 g1 = *(const float4*)(ap + 20);
      FragBF a;
      a.u[0] = cvt_pk_bf16(f0.x, f0.y); a.u[1] = cvt_pk_bf16(f0.z, f0.w);
      a.u[2] = cvt_pk_bf16(f1.x, f1.y); a.u[3] = cvt_pk_bf16(f1.z, f1.w);
      a.u[4] = cvt_pk_bf16(g0.x, g0.y); a.u[5] = cvt_pk_bf16(g0.z, g0.w);
      a.u[6] = cvt_pk_bf16(g1.x, g1.y); a.u[7] = cvt_pk_bf16(g1.z, g1.w);
      acc = __builtin_amdgcn_wmma_f32_16x16x32_bf16(false, a.bf, false, bfr.bf,
                                                    (short)0, acc, false, false);
    }
    // Diagonal recurrence: h = |u_t + hh*h|
#pragma unroll
    for (int r = 0; r < 8; ++r) hstate[r] = fabsf(fmaf(hhv, hstate[r], acc[r]));
  }

  // h_final[b, h] (fp32) to workspace. C layout: m = r + 8*hf, n = ln16.
#pragma unroll
  for (int r = 0; r < 8; ++r) {
    const int m = r + 8 * hf;
    hfin[(size_t)(b0 + m) * H_ + h0 + wave * 16 + ln16] = hstate[r];
  }
}

// ---------------------------------------------------------------------------
// Phase 2: Y = h_final @ W_ho^T + b_ho   (64 x 512 x K=2048)
// Grid (O/16, B/16), 1 wave per 16x16 tile, fragments from global fp32.
// ---------------------------------------------------------------------------
__global__ void __launch_bounds__(32)
diagnet_out(const float* __restrict__ hfin, const float* __restrict__ W_ho,
            const float* __restrict__ b_ho, float* __restrict__ Y) {
  const int lane = threadIdx.x & 31;
  const int ln16 = lane & 15;
  const int hf   = lane >> 4;
  const int o0   = blockIdx.x * 16;
  const int b0   = blockIdx.y * 16;

  const float* arow = hfin + (size_t)(b0 + ln16) * H_;
  const float* brow = W_ho + (size_t)(o0 + ln16) * H_;

  v8f acc = {};
#pragma unroll 4
  for (int kb = 0; kb < H_ / 32; ++kb) {
    FragBF a, b;
    const float4* a4 = (const float4*)(arow + kb * 32 + 8 * hf);
    float4 f0 = a4[0], f1 = a4[1];
    float4 g0 = a4[4], g1 = a4[5];
    a.u[0] = cvt_pk_bf16(f0.x, f0.y); a.u[1] = cvt_pk_bf16(f0.z, f0.w);
    a.u[2] = cvt_pk_bf16(f1.x, f1.y); a.u[3] = cvt_pk_bf16(f1.z, f1.w);
    a.u[4] = cvt_pk_bf16(g0.x, g0.y); a.u[5] = cvt_pk_bf16(g0.z, g0.w);
    a.u[6] = cvt_pk_bf16(g1.x, g1.y); a.u[7] = cvt_pk_bf16(g1.z, g1.w);
    const float4* b4 = (const float4*)(brow + kb * 32 + 8 * hf);
    float4 w0 = b4[0], w1 = b4[1], w2 = b4[4], w3 = b4[5];
    b.u[0] = cvt_pk_bf16(w0.x, w0.y); b.u[1] = cvt_pk_bf16(w0.z, w0.w);
    b.u[2] = cvt_pk_bf16(w1.x, w1.y); b.u[3] = cvt_pk_bf16(w1.z, w1.w);
    b.u[4] = cvt_pk_bf16(w2.x, w2.y); b.u[5] = cvt_pk_bf16(w2.z, w2.w);
    b.u[6] = cvt_pk_bf16(w3.x, w3.y); b.u[7] = cvt_pk_bf16(w3.z, w3.w);
    acc = __builtin_amdgcn_wmma_f32_16x16x32_bf16(false, a.bf, false, b.bf,
                                                  (short)0, acc, false, false);
  }
  const float bias = b_ho[o0 + ln16];
#pragma unroll
  for (int r = 0; r < 8; ++r)
    Y[(size_t)(b0 + r + 8 * hf) * O_ + o0 + ln16] = acc[r] + bias;
}

// ---------------------------------------------------------------------------
extern "C" void kernel_launch(void* const* d_in, const int* in_sizes, int n_in,
                              void* d_out, int out_size, void* d_ws, size_t ws_size,
                              hipStream_t stream) {
  (void)in_sizes; (void)n_in; (void)out_size; (void)ws_size;
  const float* X    = (const float*)d_in[0];  // [S, B, I]
  const float* W_ih = (const float*)d_in[1];  // [H, I]
  const float* hh   = (const float*)d_in[2];  // [H]
  const float* W_ho = (const float*)d_in[3];  // [O, H]
  const float* b_ho = (const float*)d_in[4];  // [O]
  float* Y    = (float*)d_out;                // [B, O]
  float* hfin = (float*)d_ws;                 // [B, H] = 512 KB scratch

  dim3 g1(H_ / HT, B_ / 16);  // (64, 4) = 256 WGs
  diagnet_scan<<<g1, 64, 0, stream>>>(X, W_ih, hh, hfin);

  dim3 g2(O_ / 16, B_ / 16);  // (32, 4)
  diagnet_out<<<g2, 32, 0, stream>>>(hfin, W_ho, b_ho, Y);
}
